// UniDirectionalLSTM_using_lstmcell_31525059953342
// MI455X (gfx1250) — matrix operand order repaired and checked by reading, not compiled, not run on hardware
//
#include <hip/hip_runtime.h>

// Problem constants (fixed by the reference).
#define LSTM_B 128
#define LSTM_T 1024
#define LSTM_I 256
#define LSTM_H 256
#define HPAD   264   // padded LDS row stride (keeps 16B alignment, avoids bank aliasing)

typedef __bf16 bf16_t;
typedef __attribute__((ext_vector_type(8)))  __bf16 v8bf;
typedef __attribute__((ext_vector_type(16))) __bf16 v16bf;
typedef __attribute__((ext_vector_type(8)))  float  v8f;

static __device__ inline v16bf cat8(v8bf lo, v8bf hi) {
  v16bf r;
#pragma unroll
  for (int i = 0; i < 8; ++i) { r[i] = lo[i]; r[8 + i] = hi[i]; }
  return r;
}

static __device__ inline v8f vzero8() {
  v8f r;
#pragma unroll
  for (int i = 0; i < 8; ++i) r[i] = 0.0f;
  return r;
}

// A fragment (16x32 bf16, row-major source, row = this lane's M row):
//   lanes 0-15  hold K = k0+0..15
//   lanes 16-31 hold K = k0+16..31      (off8 = (lane<16?0:16))
static __device__ inline v16bf load_a(const bf16_t* row, int k0, int off8) {
  v8bf lo = *(const v8bf*)(row + k0 + off8);
  v8bf hi = *(const v8bf*)(row + k0 + off8 + 8);
  return cat8(lo, hi);
}

// B fragment (32x16 bf16): lane L holds column N = L%16, 16 contiguous K values
// starting at k0 + (L<16 ? 0 : 16). Weights stored [4H, K] row-major => row ptr.
static __device__ inline v16bf load_b(const bf16_t* row, int k0, int off16) {
  v8bf lo = *(const v8bf*)(row + k0 + off16);
  v8bf hi = *(const v8bf*)(row + k0 + off16 + 8);
  return cat8(lo, hi);
}

static __device__ inline v8f wmma_bf16(v16bf a, v16bf b, v8f c) {
  // 8 args: (neg_a, A, neg_b, B, c_mod, C, reuse_a, reuse_b)
  return __builtin_amdgcn_wmma_f32_16x16x32_bf16(false, a, false, b, (short)0, c,
                                                 false, false);
}

static __device__ inline float fsigmoid(float x) {
  return 1.0f / (1.0f + __expf(-x));
}
static __device__ inline float ftanh(float x) {
  x = fminf(15.0f, fmaxf(-15.0f, x));
  float e = __expf(2.0f * x);
  return (e - 1.0f) / (e + 1.0f);
}

// ---------------------------------------------------------------------------
// Persistent LSTM kernel.
// Grid: 8 blocks (one per 16-row batch group). Block: 512 threads = 16 waves.
// Wave w owns h-columns [w*16, w*16+16): it computes the 16x16 C tiles of all
// four gates (i,f,g,o) for those columns, so the cell is fused in-register.
// Weights stream from L2 (shared by all 8 WGPs, ~1MB resident) every step.
// ---------------------------------------------------------------------------
__global__ __launch_bounds__(512) void lstm_persistent(
    const bf16_t* __restrict__ xw,    // [B, T, I] bf16
    const bf16_t* __restrict__ wih,   // [4H, I]   bf16
    const bf16_t* __restrict__ whh,   // [4H, H]   bf16
    const float*  __restrict__ bsum,  // [4H] = bias_ih + bias_hh
    float* __restrict__ out)          // outputs[B,T,H] ++ hT[B,H] ++ cT[B,H]
{
  __shared__ __align__(16) __bf16 h_lds[16][HPAD];

  const int tid  = threadIdx.x;
  const int wv   = tid >> 5;          // wave 0..15
  const int lane = tid & 31;
  const int l16  = lane & 15;
  const int hlf  = lane >> 4;         // 0 or 1
  const int bg   = blockIdx.x;        // batch group
  const int brow = bg * 16 + l16;     // batch row this lane loads for A frags
  const int ncol = wv * 16 + l16;     // owned h / gate column in C tiles

  // zero initial hidden state
  for (int i = tid; i < 16 * HPAD; i += 512) ((__bf16*)h_lds)[i] = (__bf16)0.0f;

  float c_state[8];
#pragma unroll
  for (int i = 0; i < 8; ++i) c_state[i] = 0.0f;

  const float b_i = bsum[ncol];
  const float b_f = bsum[LSTM_H + ncol];
  const float b_g = bsum[2 * LSTM_H + ncol];
  const float b_o = bsum[3 * LSTM_H + ncol];

  // Per-lane B-matrix row pointers for the four gate tiles.
  const bf16_t* wpi[4];
  const bf16_t* wph[4];
#pragma unroll
  for (int g = 0; g < 4; ++g) {
    wpi[g] = wih + (size_t)(g * LSTM_H + ncol) * LSTM_I;
    wph[g] = whh + (size_t)(g * LSTM_H + ncol) * LSTM_H;
  }

  const int aoff = hlf * 16;   // A lane K sub-offset
  const int boff = hlf * 16;   // B lane K sub-offset

  __syncthreads();

  for (int t = 0; t < LSTM_T; ++t) {
    const bf16_t* xrow = xw + ((size_t)brow * LSTM_T + t) * LSTM_I;
    if (t + 1 < LSTM_T) {                    // prefetch next timestep's x row
      __builtin_prefetch(xrow + LSTM_I, 0, 0);
      __builtin_prefetch(xrow + LSTM_I + 128, 0, 0);
    }

    // Opaque zero offset, regenerated every iteration: keeps weight loads
    // inside the t-loop (no hoist/spill) WITHOUT destroying the pointers'
    // global-address-space provenance (so they lower to global_load_b128,
    // not flat_load_b128 which ties up DScnt + the LDS pipe).
    unsigned long long z = 0;
    asm volatile("" : "+v"(z));
    const bf16_t* pi0 = wpi[0] + z;
    const bf16_t* pi1 = wpi[1] + z;
    const bf16_t* pi2 = wpi[2] + z;
    const bf16_t* pi3 = wpi[3] + z;
    const bf16_t* ph0 = wph[0] + z;
    const bf16_t* ph1 = wph[1] + z;
    const bf16_t* ph2 = wph[2] + z;
    const bf16_t* ph3 = wph[3] + z;

    v8f acc_i = vzero8(), acc_f = vzero8(), acc_g = vzero8(), acc_o = vzero8();

    // ---- input projection: K over I (A from global bf16 x) ----
#pragma unroll
    for (int kt = 0; kt < LSTM_I / 32; ++kt) {
      const int k0 = kt * 32;
      v16bf a = load_a(xrow, k0, aoff);
      acc_i = wmma_bf16(a, load_b(pi0, k0, boff), acc_i);
      acc_f = wmma_bf16(a, load_b(pi1, k0, boff), acc_f);
      acc_g = wmma_bf16(a, load_b(pi2, k0, boff), acc_g);
      acc_o = wmma_bf16(a, load_b(pi3, k0, boff), acc_o);
    }
    // ---- recurrent projection: K over H (A from LDS h) ----
#pragma unroll
    for (int kt = 0; kt < LSTM_H / 32; ++kt) {
      const int k0 = kt * 32;
      v16bf a = load_a(&h_lds[l16][0], k0, aoff);
      acc_i = wmma_bf16(a, load_b(ph0, k0, boff), acc_i);
      acc_f = wmma_bf16(a, load_b(ph1, k0, boff), acc_f);
      acc_g = wmma_bf16(a, load_b(ph2, k0, boff), acc_g);
      acc_o = wmma_bf16(a, load_b(ph3, k0, boff), acc_o);
    }

    __syncthreads();  // all GEMM reads of h_lds done before overwriting h

    // ---- fused LSTM cell on in-register gate tiles ----
#pragma unroll
    for (int i = 0; i < 8; ++i) {
      const int m = i + hlf * 8;     // C tile row mapping (f32 16x16 layout)
      float ig = fsigmoid(acc_i[i] + b_i);
      float fg = fsigmoid(acc_f[i] + b_f);
      float gg = ftanh(acc_g[i] + b_g);
      float og = fsigmoid(acc_o[i] + b_o);
      float c  = fg * c_state[i] + ig * gg;
      float h  = og * ftanh(c);
      c_state[i] = c;
      h_lds[m][ncol] = (__bf16)h;
      out[((size_t)(bg * 16 + m) * LSTM_T + t) * LSTM_H + ncol] = h;
    }

    __syncthreads();  // h_lds updated before next step's GEMM
  }

  // final (h_T, c_T)
  float* hT = out + (size_t)LSTM_B * LSTM_T * LSTM_H;
  float* cT = hT + (size_t)LSTM_B * LSTM_H;
#pragma unroll
  for (int i = 0; i < 8; ++i) {
    const int m = i + hlf * 8;
    hT[(size_t)(bg * 16 + m) * LSTM_H + ncol] = (float)h_lds[m][ncol];
    cT[(size_t)(bg * 16 + m) * LSTM_H + ncol] = c_state[i];
  }
}

// ---------------------------------------------------------------------------
// Preprocessing: f32 -> bf16 conversion (vectorized), bias fusion.
// ---------------------------------------------------------------------------
__global__ void cvt_f32_to_bf16_v8(const float* __restrict__ src,
                                   bf16_t* __restrict__ dst, long n8) {
  long i = (long)blockIdx.x * blockDim.x + threadIdx.x;
  const long stride = (long)gridDim.x * blockDim.x;
  for (; i < n8; i += stride) {
    const float4 a = ((const float4*)src)[2 * i];
    const float4 b = ((const float4*)src)[2 * i + 1];
    v8bf o;
    o[0] = (__bf16)a.x; o[1] = (__bf16)a.y; o[2] = (__bf16)a.z; o[3] = (__bf16)a.w;
    o[4] = (__bf16)b.x; o[5] = (__bf16)b.y; o[6] = (__bf16)b.z; o[7] = (__bf16)b.w;
    ((v8bf*)dst)[i] = o;
  }
}

__global__ void bias_sum_k(const float* __restrict__ bi,
                           const float* __restrict__ bh,
                           float* __restrict__ bs) {
  int i = blockIdx.x * blockDim.x + threadIdx.x;
  if (i < 4 * LSTM_H) bs[i] = bi[i] + bh[i];
}

extern "C" void kernel_launch(void* const* d_in, const int* in_sizes, int n_in,
                              void* d_out, int out_size, void* d_ws, size_t ws_size,
                              hipStream_t stream) {
  const float* x   = (const float*)d_in[0];
  const float* wih = (const float*)d_in[1];
  const float* whh = (const float*)d_in[2];
  const float* bih = (const float*)d_in[3];
  const float* bhh = (const float*)d_in[4];
  float* out = (float*)d_out;

  char* ws = (char*)d_ws;
  const size_t xN = (size_t)LSTM_B * LSTM_T * LSTM_I;  // 33,554,432 elems
  const size_t wN = (size_t)4 * LSTM_H * LSTM_I;       // 262,144 elems each

  bf16_t* xw   = (bf16_t*)ws;                               // 64 MB
  bf16_t* wihb = (bf16_t*)(ws + xN * 2);                    // 512 KB
  bf16_t* whhb = (bf16_t*)(ws + xN * 2 + wN * 2);           // 512 KB
  float*  bsum = (float*) (ws + xN * 2 + 2 * wN * 2);       // 4 KB

  cvt_f32_to_bf16_v8<<<2048, 256, 0, stream>>>(x, xw, (long)(xN / 8));
  cvt_f32_to_bf16_v8<<<256, 256, 0, stream>>>(wih, wihb, (long)(wN / 8));
  cvt_f32_to_bf16_v8<<<256, 256, 0, stream>>>(whh, whhb, (long)(wN / 8));
  bias_sum_k<<<4, 256, 0, stream>>>(bih, bhh, bsum);

  lstm_persistent<<<LSTM_B / 16, 512, 0, stream>>>(xw, wihb, whhb, bsum, out);
}
